// TopK_Selector_36739150250362
// MI455X (gfx1250) — matrix operand order repaired and checked by reading, not compiled, not run on hardware
//
#include <hip/hip_runtime.h>

// ---------------------------------------------------------------------------
// Types for CDNA5 WMMA
// ---------------------------------------------------------------------------
typedef float v8f  __attribute__((ext_vector_type(8)));
typedef __bf16 v16bf __attribute__((ext_vector_type(16)));
typedef unsigned short v16us __attribute__((ext_vector_type(16)));

__device__ __forceinline__ unsigned short f2bf(float f) {
    // round-to-nearest-even f32 -> bf16
    unsigned u = __float_as_uint(f);
    u += 0x7FFFu + ((u >> 16) & 1u);
    return (unsigned short)(u >> 16);
}

// B 32x16 bf16 fragment: lane = col N, lanes 0-15 hold K=kw..kw+15 (contiguous),
// lanes 16-31 hold K=kw+16..kw+31 (kw already includes half*16).
__device__ __forceinline__ v16bf load_bfrag(const float* __restrict__ p) {
    float4 b0 = *(const float4*)(p);
    float4 b1 = *(const float4*)(p + 4);
    float4 b2 = *(const float4*)(p + 8);
    float4 b3 = *(const float4*)(p + 12);
    v16us u;
    u[0]  = f2bf(b0.x); u[1]  = f2bf(b0.y); u[2]  = f2bf(b0.z); u[3]  = f2bf(b0.w);
    u[4]  = f2bf(b1.x); u[5]  = f2bf(b1.y); u[6]  = f2bf(b1.z); u[7]  = f2bf(b1.w);
    u[8]  = f2bf(b2.x); u[9]  = f2bf(b2.y); u[10] = f2bf(b2.z); u[11] = f2bf(b2.w);
    u[12] = f2bf(b3.x); u[13] = f2bf(b3.y); u[14] = f2bf(b3.z); u[15] = f2bf(b3.w);
    return __builtin_bit_cast(v16bf, u);
}

// ---------------------------------------------------------------------------
// Generic GEMM:  C[m,n] = sum_k A[m,k] * W[n,k]  + bias[n]   (C = A @ W^T + b)
// One wave computes a 16x64 tile: 1 A fragment reused by 4 B fragments ->
// 4 x v_wmma_f32_16x16x32_bf16 per 32-K step (bf16 in, f32 accumulate).
// Output row remap (shift-based, no idiv):
//   orow = rowOff + (m >> grpShift)*grpOut + (m & ((1<<grpShift)-1))
// ---------------------------------------------------------------------------
__global__ void gemm_wmma_bf16(const float* __restrict__ A, long long lda,
                               const float* __restrict__ W, int ldw,
                               const float* __restrict__ bias,
                               float* __restrict__ C, int ldc,
                               int Kd, int grpShift, int grpOut, int rowOff,
                               int relu)
{
    const int bm   = blockIdx.x * 16;
    const int bn   = blockIdx.y * 64;
    const int lane = threadIdx.x;        // 0..31
    const int half = lane >> 4;          // 0 or 1
    const int lm   = lane & 15;

    v8f acc[4] = {};
    const float* arow = A + (long long)(bm + lm) * lda;   // A: lane = row M
    const float* w0   = W + (long long)(bn + lm) * ldw;   // B: lane = col N
    const float* w1   = w0 + 16LL * ldw;
    const float* w2   = w0 + 32LL * ldw;
    const float* w3   = w0 + 48LL * ldw;

    for (int kb = 0; kb < Kd; kb += 32) {
        // A 16x32 bf16: per lane, one contiguous 16-float K run (K = kb + 16*half + s), matching B.
        const int ka = kb + (half << 4);
        float4 a0 = *(const float4*)(arow + ka);
        float4 a1 = *(const float4*)(arow + ka + 4);
        float4 a2 = *(const float4*)(arow + ka + 8);
        float4 a3 = *(const float4*)(arow + ka + 12);
        v16us ua;
        ua[0]  = f2bf(a0.x); ua[1]  = f2bf(a0.y); ua[2]  = f2bf(a0.z); ua[3]  = f2bf(a0.w);
        ua[4]  = f2bf(a1.x); ua[5]  = f2bf(a1.y); ua[6]  = f2bf(a1.z); ua[7]  = f2bf(a1.w);
        ua[8]  = f2bf(a2.x); ua[9]  = f2bf(a2.y); ua[10] = f2bf(a2.z); ua[11] = f2bf(a2.w);
        ua[12] = f2bf(a3.x); ua[13] = f2bf(a3.y); ua[14] = f2bf(a3.z); ua[15] = f2bf(a3.w);
        v16bf abf = __builtin_bit_cast(v16bf, ua);

        const int kw = kb + (half << 4);
        acc[0] = __builtin_amdgcn_wmma_f32_16x16x32_bf16(
            false, abf, false, load_bfrag(w0 + kw), (short)0, acc[0], false, false);
        acc[1] = __builtin_amdgcn_wmma_f32_16x16x32_bf16(
            false, abf, false, load_bfrag(w1 + kw), (short)0, acc[1], false, false);
        acc[2] = __builtin_amdgcn_wmma_f32_16x16x32_bf16(
            false, abf, false, load_bfrag(w2 + kw), (short)0, acc[2], false, false);
        acc[3] = __builtin_amdgcn_wmma_f32_16x16x32_bf16(
            false, abf, false, load_bfrag(w3 + kw), (short)0, acc[3], false, false);
    }

    const int grpMask = (1 << grpShift) - 1;
#pragma unroll
    for (int j = 0; j < 4; ++j) {
        const int n  = bn + j * 16 + lm;
        const float bvv = bias[n];
#pragma unroll
        for (int v = 0; v < 8; ++v) {
            int m    = bm + v + (half << 3);        // C/D layout: M = v + 8*half
            int orow = rowOff + ((m >> grpShift) * grpOut) + (m & grpMask);
            float val = acc[j][v] + bvv;
            if (relu) val = fmaxf(val, 0.0f);
            C[(long long)orow * ldc + n] = val;
        }
    }
}

// ---------------------------------------------------------------------------
// x = x*sqrt(D) + mod_emb[l<32 ? 0 : 3]
// ---------------------------------------------------------------------------
__global__ void embed_mod_kernel(float* __restrict__ x, const float* __restrict__ mod)
{
    int i = blockIdx.x * 256 + threadIdx.x;   // 768*256 total
    int n = i & 255;
    int l = (i >> 8) % 96;
    float m = (l < 32) ? mod[n] : mod[3 * 256 + n];
    x[i] = x[i] * 16.0f + m;
}

// ---------------------------------------------------------------------------
// Attention for one (b,h): L=96, dh=64. qkv rows are [q(256)|k(256)|v(256)].
// Block = 96 threads; thread t owns query row t.
// ---------------------------------------------------------------------------
__global__ void attn_kernel(const float* __restrict__ qkv, float* __restrict__ o)
{
    __shared__ float ksh[96 * 64];   // 24 KB
    __shared__ float ssh[96 * 96];   // 36 KB
    const int b = blockIdx.x >> 2;
    const int h = blockIdx.x & 3;
    const int t = threadIdx.x;       // 0..95
    const float* base = qkv + (long long)b * 96 * 768;

    for (int i = t; i < 96 * 64; i += 96) {
        int l = i >> 6, d = i & 63;
        ksh[i] = base[l * 768 + 256 + h * 64 + d];
    }
    __syncthreads();

    float qr[64];
#pragma unroll
    for (int d = 0; d < 64; ++d) qr[d] = base[t * 768 + h * 64 + d];

    float mx = -1e30f;
    for (int j = 0; j < 96; ++j) {
        float s = 0.0f;
#pragma unroll 8
        for (int d = 0; d < 64; ++d) s += qr[d] * ksh[j * 64 + d];
        s *= 0.125f;                 // 1/sqrt(64)
        ssh[t * 96 + j] = s;
        mx = fmaxf(mx, s);
    }
    float sum = 0.0f;
    for (int j = 0; j < 96; ++j) {
        float e = __expf(ssh[t * 96 + j] - mx);
        ssh[t * 96 + j] = e;
        sum += e;
    }
    float inv = 1.0f / sum;

    for (int d = 0; d < 64; ++d) {
        float a = 0.0f;
        for (int j = 0; j < 96; ++j)
            a += ssh[t * 96 + j] * base[j * 768 + 512 + h * 64 + d];
        o[((long long)b * 96 + t) * 256 + h * 64 + d] = a * inv;
    }
}

// ---------------------------------------------------------------------------
// x = LayerNorm(x + r) * s + bta    (in place on x; one row per block)
// ---------------------------------------------------------------------------
__global__ void add_ln_kernel(float* __restrict__ x, const float* __restrict__ r,
                              const float* __restrict__ s, const float* __restrict__ bta)
{
    __shared__ float red[256];
    const int row = blockIdx.x;
    const int t   = threadIdx.x;
    float v = x[row * 256 + t] + r[row * 256 + t];
    red[t] = v;
    __syncthreads();
    for (int o = 128; o > 0; o >>= 1) { if (t < o) red[t] += red[t + o]; __syncthreads(); }
    float mu = red[0] * (1.0f / 256.0f);
    __syncthreads();
    float d = v - mu;
    red[t] = d * d;
    __syncthreads();
    for (int o = 128; o > 0; o >>= 1) { if (t < o) red[t] += red[t + o]; __syncthreads(); }
    float var = red[0] * (1.0f / 256.0f);
    x[row * 256 + t] = d * __frsqrt_rn(var + 1e-5f) * s[t] + bta[t];
}

// ---------------------------------------------------------------------------
// logits[b,lv] = x[b, 32+lv, :] . Wlog + blog
// ---------------------------------------------------------------------------
__global__ void logits_kernel(const float* __restrict__ x, const float* __restrict__ wl,
                              const float* __restrict__ bl, float* __restrict__ lg)
{
    __shared__ float red[64];
    const int blk = blockIdx.x;          // 0..511
    const int b = blk >> 6, lv = blk & 63;
    const int row = b * 96 + 32 + lv;
    const int t = threadIdx.x;           // 64 threads
    float a = 0.0f;
    for (int j = t; j < 256; j += 64) a += x[row * 256 + j] * wl[j];
    red[t] = a;
    __syncthreads();
    for (int o = 32; o > 0; o >>= 1) { if (t < o) red[t] += red[t + o]; __syncthreads(); }
    if (t == 0) lg[blk] = red[0] + bl[0];
}

__global__ void zero_kernel(float* __restrict__ p, int n)
{
    int i = blockIdx.x * 256 + threadIdx.x;
    if (i < n) p[i] = 0.0f;
}

// ---------------------------------------------------------------------------
// Perturbed top-k indicators: one thread per (b, sample). K=4 over 64 logits.
// 4-pass argmax with exclusion mask (first occurrence wins ties, like top_k),
// indices sorted ascending, atomicAdd 1/NUM_SAMPLES.
// ---------------------------------------------------------------------------
__global__ void topk_kernel(const float* __restrict__ lg, const float* __restrict__ noise,
                            float* __restrict__ ind)
{
    const int i = blockIdx.x * 256 + threadIdx.x;
    if (i >= 8 * 1000) return;
    const int b = i / 1000, s = i % 1000;
    const float* L = lg + b * 64;
    const float* N = noise + ((long long)b * 1000 + s) * 64;
    unsigned long long taken = 0ull;
    int idx[4];
#pragma unroll
    for (int kk = 0; kk < 4; ++kk) {
        float best = -3.4e38f; int bi = 0;
        for (int d = 0; d < 64; ++d) {
            if (taken & (1ull << d)) continue;
            float v = L[d] + N[d] * 0.05f;
            if (v > best) { best = v; bi = d; }
        }
        taken |= (1ull << bi);
        idx[kk] = bi;
    }
#pragma unroll
    for (int a = 0; a < 3; ++a)
#pragma unroll
        for (int c = 0; c < 3 - a; ++c)
            if (idx[c] > idx[c + 1]) { int tmp = idx[c]; idx[c] = idx[c + 1]; idx[c + 1] = tmp; }
#pragma unroll
    for (int kk = 0; kk < 4; ++kk)
        atomicAdd(&ind[(b * 4 + kk) * 64 + idx[kk]], 1.0f / 1000.0f);
}

// ---------------------------------------------------------------------------
// out[b,k,n,c] = sum_d ind[b,k,d] * x_vis[b,d,n,c]   -- streaming, HBM bound.
// Reads the 741 MB x_vis exactly once; 4 outputs per element.
// ---------------------------------------------------------------------------
__global__ void select_kernel(const float* __restrict__ xv, const float* __restrict__ ind,
                              float* __restrict__ out)
{
    __shared__ float ish[256];
    const int n = blockIdx.x;   // 0..256
    const int b = blockIdx.y;   // 0..7
    const int t = threadIdx.x;  // 256
    ish[t] = ind[b * 256 + t];
    __syncthreads();

    const long long dstride = 257LL * 1408LL;
    for (int c = t; c < 1408; c += 256) {
        long long base = ((long long)b * 64 * 257 + n) * 1408 + c;
        __builtin_prefetch((const void*)(xv + base + 256), 0, 1);  // next c-chunk
        float a0 = 0.f, a1 = 0.f, a2 = 0.f, a3 = 0.f;
        for (int d = 0; d < 64; ++d) {
            float v = xv[base + (long long)d * dstride];
            a0 += ish[d] * v;
            a1 += ish[64 + d] * v;
            a2 += ish[128 + d] * v;
            a3 += ish[192 + d] * v;
        }
        long long ob = ((long long)(b * 4) * 257 + n) * 1408 + c;
        out[ob]                      = a0;
        out[ob + 1 * 257LL * 1408LL] = a1;
        out[ob + 2 * 257LL * 1408LL] = a2;
        out[ob + 3 * 257LL * 1408LL] = a3;
    }
}

// ---------------------------------------------------------------------------
// Host launcher
// ---------------------------------------------------------------------------
extern "C" void kernel_launch(void* const* d_in, const int* in_sizes, int n_in,
                              void* d_out, int out_size, void* d_ws, size_t ws_size,
                              hipStream_t stream)
{
    const float* x_vis  = (const float*)d_in[0];   // (8,64,257,1408)
    const float* x_txt  = (const float*)d_in[1];   // (8,32,2048)
    const float* Wt     = (const float*)d_in[2];   // (512,2048)
    const float* bt     = (const float*)d_in[3];
    const float* Wv     = (const float*)d_in[4];   // (512,1408)
    const float* bv     = (const float*)d_in[5];
    const float* We     = (const float*)d_in[6];   // (256,512)
    const float* be     = (const float*)d_in[7];
    const float* mod    = (const float*)d_in[8];   // (5,256)
    const float* Wqkv   = (const float*)d_in[9];   // (6,768,256)
    const float* bqkv   = (const float*)d_in[10];
    const float* Wo     = (const float*)d_in[11];  // (6,256,256)
    const float* bo     = (const float*)d_in[12];
    const float* ln1s   = (const float*)d_in[13];
    const float* ln1b   = (const float*)d_in[14];
    const float* W1     = (const float*)d_in[15];  // (6,256,256)
    const float* b1     = (const float*)d_in[16];
    const float* W2     = (const float*)d_in[17];  // (6,256,256)
    const float* b2     = (const float*)d_in[18];
    const float* ln2s   = (const float*)d_in[19];
    const float* ln2b   = (const float*)d_in[20];
    const float* Wlog   = (const float*)d_in[21];  // (256,)
    const float* blog   = (const float*)d_in[22];  // (1,)
    const float* noise  = (const float*)d_in[23];  // (8,1000,64)
    float* out = (float*)d_out;                    // (8,4,257,1408)

    float* ws   = (float*)d_ws;
    float* x512 = ws;              // 768*512  = 393216
    float* x    = ws + 393216;     // 768*256  = 196608
    float* qkv  = ws + 589824;     // 768*768  = 589824
    float* t1   = ws + 1179648;    // 768*256
    float* t2   = ws + 1376256;    // 768*256
    float* lg   = ws + 1572864;    // 512
    float* ind  = ws + 1573376;    // 8*4*64 = 2048

    // identity row remap: shift=30 -> m>>30 == 0, m & mask == m
    const int ID = 30;

    // --- input projections into x512 = concat([t, v], axis=1) -------------
    // text: (256 x 2048) @ Wt^T -> rows b*96 + lt   (group 32 -> stride 96)
    gemm_wmma_bf16<<<dim3(256 / 16, 512 / 64), 32, 0, stream>>>(
        x_txt, 2048LL, Wt, 2048, bt, x512, 512, 2048, /*shift*/5, 96, /*off*/0, 0);
    // vis CLS slice: rows stride 257*1408 -> rows b*96 + 32 + lv (group 64)
    gemm_wmma_bf16<<<dim3(512 / 16, 512 / 64), 32, 0, stream>>>(
        x_vis, 257LL * 1408LL, Wv, 1408, bv, x512, 512, 1408, 6, 96, 32, 0);
    // embed: (768 x 512) @ We^T -> x (768 x 256)
    gemm_wmma_bf16<<<dim3(768 / 16, 256 / 64), 32, 0, stream>>>(
        x512, 512LL, We, 512, be, x, 256, 512, ID, 0, 0, 0);
    embed_mod_kernel<<<768, 256, 0, stream>>>(x, mod);

    // --- 6 encoder layers -------------------------------------------------
    for (int i = 0; i < 6; ++i) {
        gemm_wmma_bf16<<<dim3(48, 12), 32, 0, stream>>>(
            x, 256LL, Wqkv + (long long)i * 768 * 256, 256, bqkv + i * 768,
            qkv, 768, 256, ID, 0, 0, 0);
        attn_kernel<<<32, 96, 0, stream>>>(qkv, t2);
        gemm_wmma_bf16<<<dim3(48, 4), 32, 0, stream>>>(
            t2, 256LL, Wo + (long long)i * 256 * 256, 256, bo + i * 256,
            t1, 256, 256, ID, 0, 0, 0);
        add_ln_kernel<<<768, 256, 0, stream>>>(x, t1, ln1s + i * 256, ln1b + i * 256);
        gemm_wmma_bf16<<<dim3(48, 4), 32, 0, stream>>>(
            x, 256LL, W1 + (long long)i * 256 * 256, 256, b1 + i * 256,
            t2, 256, 256, ID, 0, 0, /*relu*/1);
        gemm_wmma_bf16<<<dim3(48, 4), 32, 0, stream>>>(
            t2, 256LL, W2 + (long long)i * 256 * 256, 256, b2 + i * 256,
            t1, 256, 256, ID, 0, 0, 0);
        add_ln_kernel<<<768, 256, 0, stream>>>(x, t1, ln2s + i * 256, ln2b + i * 256);
    }

    // --- logits, perturbed top-k indicators, final selection --------------
    logits_kernel<<<512, 64, 0, stream>>>(x, Wlog, blog, lg);
    zero_kernel<<<8, 256, 0, stream>>>(ind, 2048);
    topk_kernel<<<32, 256, 0, stream>>>(lg, noise, ind);
    select_kernel<<<dim3(257, 8), 256, 0, stream>>>(x_vis, ind, out);
}
